// unfold_10058813407241
// MI455X (gfx1250) — matrix operand order repaired and checked by reading, not compiled, not run on hardware
//
#include <hip/hip_runtime.h>
#include <cstdint>
#include <cstddef>

typedef __attribute__((ext_vector_type(16))) _Float16 v16h;
typedef __attribute__((ext_vector_type(8)))  _Float16 v8h;
typedef __attribute__((ext_vector_type(8)))  float    v8f;

#define WMMA_F16(a,b,c) __builtin_amdgcn_wmma_f32_16x16x32_f16(false,(a),false,(b),(short)0,(c),false,false)

// ---- problem constants (from reference) ----
constexpr int cV = 50000, cIN = 300, cM = 150;
constexpr int cNL = 200000, cD = 6, cN = 40000, cK = 4;
constexpr int cB = 64, cS = 64;
constexpr int cTOTAL = 1 + cNL + cD * cN;          // 440001
constexpr int INP    = 320;                        // embed K padded to x32
constexpr int NV450  = 29;                         // valid N tiles for 450 cols
constexpr int NP450  = 32;                         // padded to 8-wave multiple
constexpr int NV150  = 10;                         // valid N tiles for 150 cols
constexpr int NP150  = 16;                         // padded
constexpr int KT320  = 10;                         // 320/32
constexpr int KT160  = 5;                          // 160/32 (150 padded)

__device__ __forceinline__ float sigmf(float x) { return 1.0f / (1.0f + __expf(-x)); }

// A fragment from a 16x32 f16 row-major LDS tile (stride 32 halves).
// ISA layout: lanes 0-15 = rows 0-15 with K 0-7 / 16-23; lanes 16-31 = K 8-15 / 24-31.
__device__ __forceinline__ v16h load_a_frag(const _Float16* sA, int lane) {
    int m  = lane & 15;
    int kb = (lane >> 4) * 8;
    v8h lo = *(const v8h*)(sA + m * 32 + kb);
    v8h hi = *(const v8h*)(sA + m * 32 + kb + 16);
    return __builtin_shufflevector(lo, hi, 0,1,2,3,4,5,6,7,8,9,10,11,12,13,14,15);
}

// B fragment from packed weights: frag-major, 32 lanes x 16 contiguous halves.
__device__ __forceinline__ v16h load_b_frag(const _Float16* pB, int frag, int lane) {
    const v8h* p = (const v8h*)(pB + (size_t)frag * 512 + lane * 16);
    return __builtin_shufflevector(p[0], p[1], 0,1,2,3,4,5,6,7,8,9,10,11,12,13,14,15);
}

// ---------------- prep kernels ----------------

__global__ void k_cvt_embed(const float* __restrict__ embed, _Float16* __restrict__ out) {
    int idx = blockIdx.x * 256 + threadIdx.x;
    if (idx >= cV * INP) return;
    int v = idx / INP, k = idx - v * INP;
    out[idx] = (k < cIN) ? (_Float16)embed[(size_t)v * cIN + k] : (_Float16)0.0f;
}

// Pack f32 weight [K x Nc] row-major into WMMA B-fragment order (zero pad past K/Nc):
// element (kt,nt,lane,i):  k = kt*32 + (lane>=16)*8 + (i&7) + (i>=8)*16,  n = nt*16 + (lane&15).
__global__ void k_pack(const float* __restrict__ W, _Float16* __restrict__ dst,
                       int K, int Nc, int NT, int total) {
    int idx = blockIdx.x * 256 + threadIdx.x;
    if (idx >= total) return;
    int frag = idx >> 9, r = idx & 511;
    int lane = r >> 4, i = r & 15;
    int kt = frag / NT, nt = frag - kt * NT;
    int k = kt * 32 + ((lane >> 4) << 3) + (i & 7) + ((i >> 3) << 4);
    int n = nt * 16 + (lane & 15);
    float v = (k < K && n < Nc) ? W[(size_t)k * Nc + n] : 0.0f;
    dst[idx] = (_Float16)v;
}

__global__ void k_zero0(float* __restrict__ c_pool, float* __restrict__ h_pool) {
    int t = threadIdx.x;
    if (t < cM) { c_pool[t] = 0.0f; h_pool[t] = 0.0f; }
}

// ---------------- leaf cell: [NL,320]f16 (gathered) @ W_leaf -> iou -> c,h ----------------

__global__ __launch_bounds__(256) void k_leaf(
    const _Float16* __restrict__ embed_h, const _Float16* __restrict__ pW_leaf,
    const float* __restrict__ b_leaf, const int* __restrict__ leaf_idx,
    float* __restrict__ c_pool, float* __restrict__ h_pool)
{
    __shared__ _Float16 sA[16 * 32];
    __shared__ float    sO[16 * 464];

    const int tid = threadIdx.x, lane = tid & 31;
    const int w = __builtin_amdgcn_readfirstlane(tid >> 5);   // wave id, provably scalar
    const int nb = blockIdx.x * 16;
    const int arow = tid >> 4, acol = (tid & 15) * 2;
    const size_t rb = (size_t)leaf_idx[nb + arow] * INP;

    v8f acc[4] = {};
    for (int kt = 0; kt < KT320; ++kt) {
        sA[arow * 32 + acol]     = embed_h[rb + kt * 32 + acol];
        sA[arow * 32 + acol + 1] = embed_h[rb + kt * 32 + acol + 1];
        __syncthreads();
        v16h a = load_a_frag(sA, lane);
#pragma unroll
        for (int t = 0; t < 4; ++t) {                         // branch-free: NP450 = 32
            v16h b = load_b_frag(pW_leaf, kt * NP450 + (w + t * 8), lane);
            acc[t] = WMMA_F16(a, b, acc[t]);
        }
        __syncthreads();
    }
#pragma unroll
    for (int t = 0; t < 4; ++t) {
        int nt = w + t * 8;
        if (nt < NV450) {                                     // scalar branch, store only
            int col = nt * 16 + (lane & 15);
            int rbse = (lane < 16) ? 0 : 8;                   // C/D layout: vgpr r -> row rbse+r
#pragma unroll
            for (int r = 0; r < 8; ++r) sO[(rbse + r) * 464 + col] = acc[t][r];
        }
    }
    __syncthreads();
    for (int e = tid; e < 16 * cM; e += 256) {
        int row = e / cM, j = e - row * cM;
        float iv = sO[row * 464 + j]          + b_leaf[j];
        float ov = sO[row * 464 + cM + j]     + b_leaf[cM + j];
        float uv = sO[row * 464 + 2 * cM + j] + b_leaf[2 * cM + j];
        float c = sigmf(iv) * tanhf(uv);
        float h = sigmf(ov) * tanhf(c);
        size_t o = (size_t)(1 + nb + row) * cM + j;
        c_pool[o] = c; h_pool[o] = h;
    }
}

// ---------------- dep step: fused iou/G1/G2 GEMMs + gated cell for 16 nodes/block ----------------

__global__ __launch_bounds__(256) void k_dep(
    const _Float16* __restrict__ embed_h,
    const _Float16* __restrict__ pW_iou, const _Float16* __restrict__ pU_iou,
    const _Float16* __restrict__ pW_f,   const _Float16* __restrict__ pU_f,
    const float* __restrict__ b_iou, const float* __restrict__ b_f,
    const int* __restrict__ word_idx, const int* __restrict__ child_idx,
    float* __restrict__ c_pool, float* __restrict__ h_pool, int start)
{
    __shared__ _Float16 sA[16 * 32];
    __shared__ float    sIOU[16 * 464];   // xd@W_iou + h_sum@U_iou
    __shared__ float    sG1[16 * 160];    // xd@W_f
    __shared__ float    sG2[16 * 160];    // one 16-child-row tile of ch_h@U_f

    const int tid = threadIdx.x, lane = tid & 31;
    const int w = __builtin_amdgcn_readfirstlane(tid >> 5);
    const int nb = blockIdx.x * 16;
    const int arow = tid >> 4, acol = (tid & 15) * 2;

    v8f aI[4] = {}, aG1[2] = {};

    // ---- phase 1: A = embed[word_idx] (K=320) -> acc_iou (W_iou), acc_g1 (W_f)
    {
        const size_t rb = (size_t)word_idx[nb + arow] * INP;
        for (int kt = 0; kt < KT320; ++kt) {
            sA[arow * 32 + acol]     = embed_h[rb + kt * 32 + acol];
            sA[arow * 32 + acol + 1] = embed_h[rb + kt * 32 + acol + 1];
            __syncthreads();
            v16h a = load_a_frag(sA, lane);
#pragma unroll
            for (int t = 0; t < 4; ++t) {
                v16h b = load_b_frag(pW_iou, kt * NP450 + (w + t * 8), lane);
                aI[t] = WMMA_F16(a, b, aI[t]);
            }
#pragma unroll
            for (int t = 0; t < 2; ++t) {
                v16h b = load_b_frag(pW_f, kt * NP150 + (w + t * 8), lane);
                aG1[t] = WMMA_F16(a, b, aG1[t]);
            }
            __syncthreads();
        }
    }

    // ---- phase 2: A = child-sum(h) f16 (K=160) -> add into acc_iou (U_iou)
    {
        const int node = nb + arow;
        const int c0 = child_idx[node * 4 + 0], c1 = child_idx[node * 4 + 1];
        const int c2 = child_idx[node * 4 + 2], c3 = child_idx[node * 4 + 3];
        for (int kt = 0; kt < KT160; ++kt) {
#pragma unroll
            for (int cc = 0; cc < 2; ++cc) {
                int kcol = kt * 32 + acol + cc;
                float s = 0.0f;
                if (kcol < cM)
                    s = h_pool[(size_t)c0 * cM + kcol] + h_pool[(size_t)c1 * cM + kcol]
                      + h_pool[(size_t)c2 * cM + kcol] + h_pool[(size_t)c3 * cM + kcol];
                sA[arow * 32 + acol + cc] = (_Float16)s;
            }
            __syncthreads();
            v16h a = load_a_frag(sA, lane);
#pragma unroll
            for (int t = 0; t < 4; ++t) {
                v16h b = load_b_frag(pU_iou, kt * NP450 + (w + t * 8), lane);
                aI[t] = WMMA_F16(a, b, aI[t]);
            }
            __syncthreads();
        }
    }

    // park iou / g1 in LDS for the epilogue
    {
        int col16 = lane & 15;
        int rbse = (lane < 16) ? 0 : 8;
#pragma unroll
        for (int t = 0; t < 4; ++t) {
            int nt = w + t * 8;
            if (nt < NV450) {
                int col = nt * 16 + col16;
#pragma unroll
                for (int r = 0; r < 8; ++r) sIOU[(rbse + r) * 464 + col] = aI[t][r];
            }
        }
#pragma unroll
        for (int t = 0; t < 2; ++t) {
            int nt = w + t * 8;
            if (nt < NV150) {
                int col = nt * 16 + col16;
#pragma unroll
                for (int r = 0; r < 8; ++r) sG1[(rbse + r) * 160 + col] = aG1[t][r];
            }
        }
    }
    __syncthreads();

    // ---- phase 3: per 16-child-row tile rt: G2 = ch_h @ U_f, then finalize 4 nodes
    for (int rt = 0; rt < 4; ++rt) {
        const int cr = rt * 16 + arow;                      // child-row this thread stages
        const int node3 = nb + (cr >> 2), k3 = cr & 3;
        const int slot3 = child_idx[node3 * 4 + k3];
        v8f aG2[2] = {};
        for (int kt = 0; kt < KT160; ++kt) {
#pragma unroll
            for (int cc = 0; cc < 2; ++cc) {
                int kcol = kt * 32 + acol + cc;
                float v = (kcol < cM) ? h_pool[(size_t)slot3 * cM + kcol] : 0.0f;
                sA[arow * 32 + acol + cc] = (_Float16)v;
            }
            __syncthreads();
            v16h a = load_a_frag(sA, lane);
#pragma unroll
            for (int t = 0; t < 2; ++t) {
                v16h b = load_b_frag(pU_f, kt * NP150 + (w + t * 8), lane);
                aG2[t] = WMMA_F16(a, b, aG2[t]);
            }
            __syncthreads();
        }
        {
            int col16 = lane & 15;
            int rbse = (lane < 16) ? 0 : 8;
#pragma unroll
            for (int t = 0; t < 2; ++t) {
                int nt = w + t * 8;
                if (nt < NV150) {
                    int col = nt * 16 + col16;
#pragma unroll
                    for (int r = 0; r < 8; ++r) sG2[(rbse + r) * 160 + col] = aG2[t][r];
                }
            }
        }
        __syncthreads();
        // finalize the 4 nodes covered by this row tile
        for (int e = tid; e < 4 * cM; e += 256) {
            int a4 = e / cM, j = e - a4 * cM;
            int ln = rt * 4 + a4, node = nb + ln;
            float g1 = sG1[ln * 160 + j] + b_f[j];
            float csum = 0.0f;
#pragma unroll
            for (int k = 0; k < 4; ++k) {
                float fv = sigmf(g1 + sG2[(a4 * 4 + k) * 160 + j]);
                float cc = c_pool[(size_t)child_idx[node * 4 + k] * cM + j];
                csum += fv * cc;
            }
            float iv = sIOU[ln * 464 + j]          + b_iou[j];
            float ov = sIOU[ln * 464 + cM + j]     + b_iou[cM + j];
            float uv = sIOU[ln * 464 + 2 * cM + j] + b_iou[2 * cM + j];
            float c = sigmf(iv) * tanhf(uv) + csum;
            float h = sigmf(ov) * tanhf(c);
            size_t o = (size_t)(start + node) * cM + j;
            c_pool[o] = c; h_pool[o] = h;
        }
        __syncthreads();
    }
}

// ---------------- final context gather ----------------

__global__ void k_gather(const float* __restrict__ c_pool, const float* __restrict__ h_pool,
                         const int* __restrict__ ctx, float* __restrict__ out) {
    int idx = blockIdx.x * 256 + threadIdx.x;
    if (idx >= cB * cS * cM) return;
    int p = idx / cM, j = idx - p * cM;
    size_t src = (size_t)ctx[p] * cM + j;
    out[idx] = c_pool[src];
    out[(size_t)cB * cS * cM + idx] = h_pool[src];
}

// ---------------- host ----------------

extern "C" void kernel_launch(void* const* d_in, const int* in_sizes, int n_in,
                              void* d_out, int out_size, void* d_ws, size_t ws_size,
                              hipStream_t stream) {
    const float* embed  = (const float*)d_in[0];
    const float* W_leaf = (const float*)d_in[1];
    const float* b_leaf = (const float*)d_in[2];
    const float* W_iou  = (const float*)d_in[3];
    const float* U_iou  = (const float*)d_in[4];
    const float* b_iou  = (const float*)d_in[5];
    const float* W_f    = (const float*)d_in[6];
    const float* U_f    = (const float*)d_in[7];
    const float* b_f    = (const float*)d_in[8];
    const int* leaf_idx  = (const int*)d_in[9];
    const int* word_idx  = (const int*)d_in[10];
    const int* child_idx = (const int*)d_in[11];
    const int* ctx_idx   = (const int*)d_in[12];

    char* ws = (char*)d_ws;
    size_t off = 0;
    auto take = [&](size_t bytes) -> char* {
        char* p = ws + off;
        off += (bytes + 255) & ~(size_t)255;
        return p;
    };
    float*    c_pool  = (float*)take((size_t)cTOTAL * cM * sizeof(float));
    float*    h_pool  = (float*)take((size_t)cTOTAL * cM * sizeof(float));
    _Float16* embed_h = (_Float16*)take((size_t)cV * INP * sizeof(_Float16));
    _Float16* pW_leaf = (_Float16*)take((size_t)KT320 * NP450 * 512 * 2);
    _Float16* pW_iou  = (_Float16*)take((size_t)KT320 * NP450 * 512 * 2);
    _Float16* pU_iou  = (_Float16*)take((size_t)KT160 * NP450 * 512 * 2);
    _Float16* pW_f    = (_Float16*)take((size_t)KT320 * NP150 * 512 * 2);
    _Float16* pU_f    = (_Float16*)take((size_t)KT160 * NP150 * 512 * 2);
    if (off > ws_size) return;

    k_cvt_embed<<<(cV * INP + 255) / 256, 256, 0, stream>>>(embed, embed_h);

    k_pack<<<(KT320 * NP450 * 512 + 255) / 256, 256, 0, stream>>>(W_leaf, pW_leaf, cIN, 3 * cM, NP450, KT320 * NP450 * 512);
    k_pack<<<(KT320 * NP450 * 512 + 255) / 256, 256, 0, stream>>>(W_iou,  pW_iou,  cIN, 3 * cM, NP450, KT320 * NP450 * 512);
    k_pack<<<(KT160 * NP450 * 512 + 255) / 256, 256, 0, stream>>>(U_iou,  pU_iou,  cM,  3 * cM, NP450, KT160 * NP450 * 512);
    k_pack<<<(KT320 * NP150 * 512 + 255) / 256, 256, 0, stream>>>(W_f,    pW_f,    cIN, cM,     NP150, KT320 * NP150 * 512);
    k_pack<<<(KT160 * NP150 * 512 + 255) / 256, 256, 0, stream>>>(U_f,    pU_f,    cM,  cM,     NP150, KT160 * NP150 * 512);

    k_zero0<<<1, 256, 0, stream>>>(c_pool, h_pool);

    k_leaf<<<cNL / 16, 256, 0, stream>>>(embed_h, pW_leaf, b_leaf, leaf_idx, c_pool, h_pool);

    for (int d = 0; d < cD; ++d) {
        k_dep<<<cN / 16, 256, 0, stream>>>(embed_h, pW_iou, pU_iou, pW_f, pU_f,
                                           b_iou, b_f,
                                           word_idx + (size_t)d * cN,
                                           child_idx + (size_t)d * cN * cK,
                                           c_pool, h_pool, 1 + cNL + d * cN);
    }

    k_gather<<<(cB * cS * cM + 255) / 256, 256, 0, stream>>>(c_pool, h_pool, ctx_idx, (float*)d_out);
}
